// Net_dnc_86595130622420
// MI455X (gfx1250) — matrix operand order repaired and checked by reading, not compiled, not run on hardware
//
#include <hip/hip_runtime.h>
#include <hip/hip_bf16.h>
#include <math.h>

// ---------------- constants from the reference ----------------
#define NUM_SYMBOLS 1000000
#define EMB   20
#define HID   20
#define G4    80        // 4*HID LSTM gates
#define CTRL  64        // DNC controller hidden
#define CG    256       // 4*CTRL
#define CIN   84        // HID + READ_VEC controller input
#define MEMH  16
#define MEMW  16
#define NREAD 4
#define RVEC  64        // NREAD*MEMW
#define SEQ   8192
#define EPSV  1e-6f
#define CLIPV 20.0f

typedef __attribute__((ext_vector_type(16))) _Float16 v16h;
typedef __attribute__((ext_vector_type(8)))  float    v8f;

__device__ __forceinline__ float sigf(float x) { return 1.0f / (1.0f + __expf(-x)); }
__device__ __forceinline__ float softplusf(float x) {
  return x > 0.0f ? x + log1pf(__expf(-x)) : log1pf(__expf(x));
}

// =====================================================================
// Kernel A: parallel input projection via WMMA.
// Apre[t, g] = (emb[x_t-1]*mask) . W_ih[g, :] + b_ih[g] + b_hh[g]
// Tiled 16 timesteps x 16 gates, K = 32 (EMB=20 zero-padded).
// One wave per tile -> one v_wmma_f32_16x16x32_f16 each.
// =====================================================================
__global__ void __launch_bounds__(32)
precompute_gates(const int* __restrict__ x,
                 const float* __restrict__ emb,
                 const float* __restrict__ wih,   // [80,20]
                 const float* __restrict__ bih,   // [80]
                 const float* __restrict__ bhh,   // [80]
                 float* __restrict__ Apre) {      // [SEQ,80]
  const int lane = threadIdx.x;       // 0..31, one wave
  const int hi   = lane >> 4;         // lane half selects K range
  const int l    = lane & 15;
  const int t    = blockIdx.x * 16 + l;   // timestep row (A layout: M = lane&15)
  const int gc   = blockIdx.y * 16 + l;   // gate column (B/D layout: N = lane&15)

  // ---- A tile (f16 16x32): lanes 0-15 hold K 0-7 / 16-23, lanes 16-31 hold K 8-15 / 24-31
  int xi = x[t] - 1;
  float mask = (xi != NUM_SYMBOLS - 1) ? 1.0f : 0.0f;
  const float* er = emb + (size_t)xi * EMB;
  v16h a;
#pragma unroll
  for (int i = 0; i < 16; ++i) {
    int k = (i < 8 ? i : i + 8) + (hi ? 8 : 0);
    a[i] = (k < EMB) ? (_Float16)(er[k] * mask) : (_Float16)0.0f;
  }

  // ---- B tile (f16 32x16): B[k][n] = W_ih[n][k]; lanes 0-15 K 0-7 / 16-23, lanes 16-31 K 8-15 / 24-31
  const int koff = hi ? 8 : 0;
  v16h b;
#pragma unroll
  for (int i = 0; i < 16; ++i) {
    int k = koff + (i < 8 ? i : i + 8);
    b[i] = (k < EMB) ? (_Float16)wih[gc * EMB + k] : (_Float16)0.0f;
  }

  v8f acc = {};
  acc = __builtin_amdgcn_wmma_f32_16x16x32_f16(
      /*neg_a=*/false, a, /*neg_b=*/false, b,
      /*c_mod=*/(short)0, acc, /*reuse_a=*/false, /*reuse_b=*/false);

  // ---- D (f32 16x16): VGPR v holds M = v (+8 for upper lanes), N = lane&15
#pragma unroll
  for (int v = 0; v < 8; ++v) {
    int tt = blockIdx.x * 16 + v + (hi ? 8 : 0);
    Apre[(size_t)tt * G4 + gc] = acc[v] + bih[gc] + bhh[gc];
  }
}

// =====================================================================
// Kernel B: the unavoidable serial recurrence. One wave, W_hh in LDS
// (stride 21 to avoid bank conflicts), one-step-ahead prefetch of Apre
// (L2-resident) to overlap global latency with the step math.
// =====================================================================
__global__ void __launch_bounds__(32)
lstm_scan(const float* __restrict__ Apre,
          const float* __restrict__ whh,     // [80,20]
          float* __restrict__ x4out) {       // [20] = sum_t h_t
  __shared__ float Wh[G4 * 21];
  __shared__ float hs[HID], cs[HID], hsum[HID];
  __shared__ float gate[G4];
  const int j = threadIdx.x;   // 32 threads

  for (int idx = j; idx < G4 * HID; idx += 32) {
    int r = idx / HID, k = idx - r * HID;
    Wh[r * 21 + k] = whh[idx];
  }
  if (j < HID) { hs[j] = 0.0f; cs[j] = 0.0f; hsum[j] = 0.0f; }
  __syncthreads();

  // rows handled per thread: j, j+32, and (j<16) j+64
  float a0 = Apre[j], a1 = Apre[j + 32];
  float a2 = (j < 16) ? Apre[j + 64] : 0.0f;

  for (int t = 0; t < SEQ; ++t) {
    // prefetch next step's pre-activations (independent of h-chain)
    const float* nx = Apre + (size_t)(t + 1) * G4;
    float n0 = 0.0f, n1 = 0.0f, n2 = 0.0f;
    if (t + 1 < SEQ) {
      n0 = nx[j]; n1 = nx[j + 32];
      if (j < 16) n2 = nx[j + 64];
    }

    float g0 = a0, g1 = a1, g2 = a2;
#pragma unroll
    for (int k = 0; k < HID; ++k) {
      float hk = hs[k];
      g0 = fmaf(Wh[j * 21 + k], hk, g0);
      g1 = fmaf(Wh[(j + 32) * 21 + k], hk, g1);
      if (j < 16) g2 = fmaf(Wh[(j + 64) * 21 + k], hk, g2);
    }
    gate[j] = g0; gate[j + 32] = g1;
    if (j < 16) gate[j + 64] = g2;
    __syncthreads();

    if (j < HID) {
      float ig = gate[j], fg = gate[HID + j], gg = gate[2 * HID + j], og = gate[3 * HID + j];
      float c = sigf(fg) * cs[j] + sigf(ig) * tanhf(gg);
      cs[j] = c;
      float h = sigf(og) * tanhf(c);
      hs[j] = h;
      hsum[j] += h;
    }
    __syncthreads();
    a0 = n0; a1 = n1; a2 = n2;
  }
  if (j < HID) x4out[j] = hsum[j];
}

// =====================================================================
// Kernel C: DNC step from freshly-reset state + output head.
// Fresh state makes usage==0, link==0, Mn uniform -> both content
// softmaxes are exactly 1/16, argsort is identity, fwd=bwd=0.
// =====================================================================
struct DncParams {
  const float *c_wih, *c_bih, *c_bhh;
  const float *wkey_w, *wkey_b;
  const float *wbeta_w, *wbeta_b;
  const float *werase_w, *werase_b;
  const float *wadd_w, *wadd_b;
  const float *wagate_w, *wagate_b;
  const float *wwgate_w, *wwgate_b;
  const float *rkey_w, *rkey_b;
  const float *rbeta_w, *rbeta_b;
  const float *rfree_w, *rfree_b;
  const float *rmode_w, *rmode_b;
  const float *h2o_w, *h2o_b;
  const float *lin_w, *lin_b;
  const float *obj_w, *obj_b;
};

__global__ void __launch_bounds__(256)
dnc_head(DncParams p, const float* __restrict__ x4in, float* __restrict__ out) {
  __shared__ float x4s[HID];
  __shared__ float g[CG];
  __shared__ float hc[CTRL];
  __shared__ float wks[MEMW], ers[MEMW], ads[MEMW];
  __shared__ float rks[NREAD * MEMW], rbet[NREAD];
  __shared__ float mraw[NREAD * 3], m1[NREAD];
  __shared__ float sc3[3];                 // {softplus wbeta, g_alloc, g_write}
  __shared__ float wcs[MEMH], rcs[NREAD * MEMH], wws[MEMH];
  __shared__ float Mn[MEMH * MEMW];
  __shared__ float rv[RVEC];
  __shared__ float o1[HID], x5[HID];
  const int tid = threadIdx.x;             // 256 threads

  if (tid < HID) x4s[tid] = x4in[tid];
  __syncthreads();

  // ---- controller LSTMCell: h0=c0=0, read_prev=0 -> only first HID inputs matter
  {
    float acc = p.c_bih[tid] + p.c_bhh[tid];
    const float* wr = p.c_wih + (size_t)tid * CIN;
    for (int k = 0; k < HID; ++k) acc = fmaf(wr[k], x4s[k], acc);
    g[tid] = acc;
  }
  __syncthreads();
  if (tid < CTRL) {
    float ig = g[tid], gg = g[2 * CTRL + tid], og = g[3 * CTRL + tid];
    float c = sigf(ig) * tanhf(gg);        // + sig(f)*c0 with c0==0
    float h = sigf(og) * tanhf(c);
    hc[tid] = fminf(fmaxf(h, -CLIPV), CLIPV);
  }
  __syncthreads();

  // ---- interface heads (each thread = one output row)
  auto dot64 = [&](const float* w) -> float {
    float a = 0.0f;
    for (int k = 0; k < CTRL; ++k) a = fmaf(w[k], hc[k], a);
    return a;
  };
  if (tid < 16)        { wks[tid] = tanhf(dot64(p.wkey_w + tid * CTRL) + p.wkey_b[tid]); }
  else if (tid < 32)   { int r = tid - 16;  ers[r]  = sigf (dot64(p.werase_w + r * CTRL) + p.werase_b[r]); }
  else if (tid < 48)   { int r = tid - 32;  ads[r]  = tanhf(dot64(p.wadd_w   + r * CTRL) + p.wadd_b[r]);   }
  else if (tid < 112)  { int r = tid - 48;  rks[r]  = tanhf(dot64(p.rkey_w   + r * CTRL) + p.rkey_b[r]);   }
  else if (tid < 116)  { int r = tid - 112; rbet[r] = softplusf(dot64(p.rbeta_w + r * CTRL) + p.rbeta_b[r]); }
  else if (tid < 128)  { int r = tid - 116; mraw[r] = dot64(p.rmode_w + r * CTRL) + p.rmode_b[r]; }
  else if (tid == 128) { sc3[0] = softplusf(dot64(p.wbeta_w)  + p.wbeta_b[0]);  }
  else if (tid == 129) { sc3[1] = sigf     (dot64(p.wagate_w) + p.wagate_b[0]); }
  else if (tid == 130) { sc3[2] = sigf     (dot64(p.wwgate_w) + p.wwgate_b[0]); }
  __syncthreads();

  if (tid == 0) {
    // M0 rows uniform (1e-6) -> normalized rows uniform -> content scores equal
    // per row -> softmax is exactly 1/16 regardless of keys/betas.
    for (int hh = 0; hh < MEMH; ++hh) wcs[hh] = 1.0f / (float)MEMH;
    for (int r = 0; r < NREAD; ++r)
      for (int hh = 0; hh < MEMH; ++hh) rcs[r * MEMH + hh] = 1.0f / (float)MEMH;

    // usage==0 (free*w_r_prev=0 -> psi=1; w_w_prev=0): u = EPS everywhere,
    // stable argsort = identity; alloc[i] = (1-EPS)*EPS^i.
    float prod = 1.0f;
    float galloc = sc3[1], gwrite = sc3[2];
    for (int i = 0; i < MEMH; ++i) {
      float alloc = (1.0f - EPSV) * prod;
      prod *= EPSV;
      wws[i] = gwrite * (galloc * alloc + (1.0f - galloc) * wcs[i]);
    }
    // read-mode softmax; link==0 -> fwd=bwd=0, only content column survives
    for (int r = 0; r < NREAD; ++r) {
      float a0 = mraw[r * 3 + 0], a1 = mraw[r * 3 + 1], a2 = mraw[r * 3 + 2];
      float mx = fmaxf(a0, fmaxf(a1, a2));
      float e0 = __expf(a0 - mx), e1 = __expf(a1 - mx), e2 = __expf(a2 - mx);
      m1[r] = e1 / (e0 + e1 + e2);
    }
  }
  __syncthreads();

  // ---- memory erase/add: M = M0*(1 - ww⊗erase) + ww⊗add   (tid covers 16x16)
  {
    int hh = tid >> 4, w = tid & 15;
    Mn[tid] = EPSV * (1.0f - wws[hh] * ers[w]) + wws[hh] * ads[w];
  }
  __syncthreads();

  // ---- read vectors: wr[r][h] = m1[r] * rc[r][h]; rv = wr @ M
  if (tid < RVEC) {
    int r = tid >> 4, w = tid & 15;
    float a = 0.0f;
    for (int hh = 0; hh < MEMH; ++hh)
      a = fmaf(m1[r] * rcs[r * MEMH + hh], Mn[hh * MEMW + w], a);
    rv[tid] = a;
  }
  __syncthreads();

  // ---- hid2out([h, read_vec]) -> o1[20]
  if (tid < HID) {
    const float* w = p.h2o_w + (size_t)tid * (CTRL + RVEC);
    float a = p.h2o_b[tid];
    for (int k = 0; k < CTRL; ++k) a = fmaf(w[k], hc[k], a);
    for (int k = 0; k < RVEC; ++k) a = fmaf(w[CTRL + k], rv[k], a);
    o1[tid] = a;
  }
  __syncthreads();

  // ---- x5 = relu(linear([x4, o1]))
  if (tid < HID) {
    const float* w = p.lin_w + (size_t)tid * (2 * HID);
    float a = p.lin_b[tid];
    for (int k = 0; k < HID; ++k) a = fmaf(w[k], x4s[k], a);
    for (int k = 0; k < HID; ++k) a = fmaf(w[HID + k], o1[k], a);
    x5[tid] = fmaxf(a, 0.0f);
  }
  __syncthreads();

  // ---- objects: [1000,20] @ x5 + b
  for (int r = tid; r < 1000; r += 256) {
    const float* w = p.obj_w + (size_t)r * HID;
    float a = p.obj_b[r];
    for (int k = 0; k < HID; ++k) a = fmaf(w[k], x5[k], a);
    out[r] = a;
  }
}

// =====================================================================
extern "C" void kernel_launch(void* const* d_in, const int* in_sizes, int n_in,
                              void* d_out, int out_size, void* d_ws, size_t ws_size,
                              hipStream_t stream) {
  (void)in_sizes; (void)n_in; (void)out_size; (void)ws_size;
  // setup_inputs() insertion order, recursively flattened:
  int i = 0;
  const int*   x     = (const int*)  d_in[i++];   // x [1,8192]
  const float* emb   = (const float*)d_in[i++];   // embedding [1e6,20]
  const float* l_wih = (const float*)d_in[i++];   // lstm.w_ih [80,20]
  const float* l_whh = (const float*)d_in[i++];   // lstm.w_hh [80,20]
  const float* l_bih = (const float*)d_in[i++];   // lstm.b_ih [80]
  const float* l_bhh = (const float*)d_in[i++];   // lstm.b_hh [80]

  DncParams p;
  p.c_wih = (const float*)d_in[i++];              // ctrl.w_ih [256,84]
  i++;                                            // ctrl.w_hh unused (h0=0)
  p.c_bih = (const float*)d_in[i++];
  p.c_bhh = (const float*)d_in[i++];
  p.wkey_w  = (const float*)d_in[i++]; p.wkey_b  = (const float*)d_in[i++];
  p.wbeta_w = (const float*)d_in[i++]; p.wbeta_b = (const float*)d_in[i++];
  p.werase_w= (const float*)d_in[i++]; p.werase_b= (const float*)d_in[i++];
  p.wadd_w  = (const float*)d_in[i++]; p.wadd_b  = (const float*)d_in[i++];
  p.wagate_w= (const float*)d_in[i++]; p.wagate_b= (const float*)d_in[i++];
  p.wwgate_w= (const float*)d_in[i++]; p.wwgate_b= (const float*)d_in[i++];
  p.rkey_w  = (const float*)d_in[i++]; p.rkey_b  = (const float*)d_in[i++];
  p.rbeta_w = (const float*)d_in[i++]; p.rbeta_b = (const float*)d_in[i++];
  p.rfree_w = (const float*)d_in[i++]; p.rfree_b = (const float*)d_in[i++];
  p.rmode_w = (const float*)d_in[i++]; p.rmode_b = (const float*)d_in[i++];
  p.h2o_w   = (const float*)d_in[i++]; p.h2o_b   = (const float*)d_in[i++];
  p.lin_w   = (const float*)d_in[i++]; p.lin_b   = (const float*)d_in[i++];
  p.obj_w   = (const float*)d_in[i++]; p.obj_b   = (const float*)d_in[i++];

  float* Apre = (float*)d_ws;                     // [SEQ,80] = 2.62 MB
  float* x4ws = Apre + (size_t)SEQ * G4;          // [20]

  precompute_gates<<<dim3(SEQ / 16, G4 / 16), 32, 0, stream>>>(
      x, emb, l_wih, l_bih, l_bhh, Apre);
  lstm_scan<<<1, 32, 0, stream>>>(Apre, l_whh, x4ws);
  dnc_head<<<1, 256, 0, stream>>>(p, x4ws, (float*)d_out);
}
